// CRFNERModel_43971875176817
// MI455X (gfx1250) — matrix operand order repaired and checked by reading, not compiled, not run on hardware
//
#include <hip/hip_runtime.h>
#include <hip/hip_bf16.h>
#include <math.h>

// ---------------------------------------------------------------------------
// Model dims (compile-time constants from the reference)
// ---------------------------------------------------------------------------
#define BB   32
#define SS   128
#define HH   768
#define LL   12
#define NHH  12
#define DHH  64
#define FFF  3072
#define TT   9
#define NTOK 4096            // B*S
#define LN_EPS 1e-12f

typedef __attribute__((ext_vector_type(16))) __bf16 v16bf;
typedef __attribute__((ext_vector_type(8)))  __bf16 v8bf;
typedef __attribute__((ext_vector_type(8)))  float  v8f;
typedef __attribute__((ext_vector_type(4)))  float  v4f;
typedef __attribute__((ext_vector_type(4)))  unsigned int v4u;
typedef __attribute__((ext_vector_type(8)))  int    v8i;
typedef __attribute__((ext_vector_type(4)))  int    v4i;

// Tensor Data Mover availability (guarded so compilation never regresses).
#if defined(__gfx1250__) && __has_builtin(__builtin_amdgcn_tensor_load_to_lds) && \
    __has_builtin(__builtin_amdgcn_s_wait_tensorcnt)
#define USE_TDM 1
#if __has_include(<hip/amd_detail/amd_gfx1250_TDM.h>)
#define TDM_6ARG 1   // therock headers -> 6-arg builtin
#endif
#endif

// ---------------------------------------------------------------------------
// WMMA fragment helpers (CDNA5 16x16x32 bf16 layouts, wave32)
//   A 16x32: lane m=L%16, hi=L/16; elems 0..7 -> K = 8*hi..8*hi+7,
//            elems 8..15 -> K = 16+8*hi .. 16+8*hi+7
//   B 32x16 (stored transposed in LDS as [N][K]): same K mapping as A
//   C/D 16x16 f32: VGPR r -> row r + 8*hi, col = L%16
// ---------------------------------------------------------------------------
__device__ __forceinline__ v16bf make_frag(const __bf16* p0, const __bf16* p1) {
    v8bf lo = *(const v8bf*)p0;
    v8bf hv = *(const v8bf*)p1;
    v16bf r;
#pragma unroll
    for (int i = 0; i < 8; ++i) { r[i] = lo[i]; r[i + 8] = hv[i]; }
    return r;
}
__device__ __forceinline__ v16bf frag_A(const __bf16* row, int hi) {
    return make_frag(row + 8 * hi, row + 16 + 8 * hi);
}
__device__ __forceinline__ v16bf frag_B(const __bf16* row, int hi) {
    return make_frag(row + 8 * hi, row + 16 + 8 * hi);
}
__device__ __forceinline__ v8f mma(v16bf a, v16bf b, v8f c) {
    return __builtin_amdgcn_wmma_f32_16x16x32_bf16(false, a, false, b, (short)0, c,
                                                   false, false);
}

#ifdef USE_TDM
// ---------------------------------------------------------------------------
// Issue one TDM 2-D tile load: 128 rows x 32 cols of bf16 from a row-major
// [rows, kdim] bf16 matrix into LDS, padding 4 DWORDs after each 16-DWORD row
// (64B row + 16B pad -> LDS row stride 40 bf16 == our [128][40] layout).
// D# built per CDNA5 ISA 8.3/8.4 (group0 128b, group1 256b), groups 2/3 zero.
// Wave-uniform; call from a single wave. Tracked by TENSORcnt.
// ---------------------------------------------------------------------------
__device__ __forceinline__ void tdm_tile_load(void* lds_dst, const __bf16* gsrc,
                                              int kdim, int rows) {
    const unsigned long long ga = (unsigned long long)(size_t)gsrc;
    const unsigned lds_addr = (unsigned)(size_t)lds_dst;  // LDS aperture low bits
    v4u g0;
    g0[0] = 1u;                                            // count=1 (valid), user mode
    g0[1] = lds_addr;                                      // lds_addr[31:0]
    g0[2] = (unsigned)(ga & 0xFFFFFFFFu);                  // global_addr[31:0]
    g0[3] = (unsigned)((ga >> 32) & 0x01FFFFFFu) | (2u << 30);  // addr[56:32], type=2
    const unsigned td0 = (unsigned)kdim;                   // tensor_dim0 (elements)
    const unsigned td1 = (unsigned)rows;                   // tensor_dim1
    const unsigned st0 = (unsigned)kdim;                   // tensor_dim0_stride
    v8i g1;
    g1[0] = (int)((1u << 16)      // data_size = 1 -> 2 bytes
                  | (1u << 20)    // pad_enable
                  | (3u << 22)    // pad_interval: 16 DWORDs
                  | (3u << 25));  // pad_amount: 4 DWORDs
    g1[1] = (int)((td0 & 0xFFFFu) << 16);                         // dim0[15:0]
    g1[2] = (int)(((td0 >> 16) & 0xFFFFu) | ((td1 & 0xFFFFu) << 16));
    g1[3] = (int)(((td1 >> 16) & 0xFFFFu) | (32u << 16));         // tile_dim0 = 32
    g1[4] = (int)128;                                             // tile_dim1 = 128
    g1[5] = (int)st0;                                             // dim0_stride[31:0]
    g1[6] = 0;                                                    // stride hi / dim1_stride
    g1[7] = 0;
    const v4i z = {0, 0, 0, 0};
#ifdef TDM_6ARG
    const v8i z8 = {0, 0, 0, 0, 0, 0, 0, 0};
    __builtin_amdgcn_tensor_load_to_lds(g0, g1, z, z, z8, 0);
#else
    __builtin_amdgcn_tensor_load_to_lds(g0, g1, z, z, 0);
#endif
}
#endif  // USE_TDM

// ---------------------------------------------------------------------------
// Weight pre-pass: fp32 W[K,N] -> bf16 Wt[N,K] (transpose via 32x32 LDS tile)
// ---------------------------------------------------------------------------
__global__ __launch_bounds__(256) void wt_kernel(const float* __restrict__ W,
                                                 __bf16* __restrict__ Wt, int K,
                                                 int N) {
    __shared__ float t[32][33];
    const int bk = blockIdx.y * 32, bn = blockIdx.x * 32;
    const int tx = threadIdx.x & 31, ty = threadIdx.x >> 5;
#pragma unroll
    for (int r = 0; r < 32; r += 8)
        t[ty + r][tx] = W[(size_t)(bk + ty + r) * N + bn + tx];
    __syncthreads();
#pragma unroll
    for (int r = 0; r < 32; r += 8)
        Wt[(size_t)(bn + ty + r) * K + bk + tx] = (__bf16)t[tx][ty + r];
}

// ---------------------------------------------------------------------------
// WMMA GEMM: out = epilogue(A_bf16[M,K] @ Bt_bf16[N,K]^T + bias[N])
// 128x128 block tile, BK=32, 256 threads (8 waves), wave tile 32x64.
// Double-buffered LDS; tiles staged by the Tensor Data Mover when available
// (wave 0 issues, TENSORcnt + workgroup barrier synchronize), else vector ld.
// MODE 0: QKV scatter (Q scaled 1/8 -> Qb[b,h,s,d], K -> Kb, V -> Vt[b,h,d,s])
// MODE 1: fp32 out    (outF[M,N])
// MODE 2: exact GELU  (outH[M,N] bf16)
// ---------------------------------------------------------------------------
template <int MODE>
__global__ __launch_bounds__(256) void gemm_kernel(
    const __bf16* __restrict__ A, const __bf16* __restrict__ Bt,
    const float* __restrict__ bias, float* __restrict__ outF,
    __bf16* __restrict__ outH, __bf16* __restrict__ Qb, __bf16* __restrict__ Kb,
    __bf16* __restrict__ Vt, int M, int N, int K) {
    __shared__ __align__(16) __bf16 As[2][128][40];
    __shared__ __align__(16) __bf16 Bs[2][128][40];

    const int tid  = threadIdx.x;
    const int lane = tid & 31, wid = tid >> 5;
    const int hi = lane >> 4, l16 = lane & 15;
    const int wm = (wid & 3) * 32, wn = (wid >> 2) * 64;
    const int blkM = blockIdx.y * 128, blkN = blockIdx.x * 128;
    const int nk = K >> 5;

    v8f acc[2][4];
#pragma unroll
    for (int i = 0; i < 2; ++i)
#pragma unroll
        for (int j = 0; j < 4; ++j)
#pragma unroll
            for (int r = 0; r < 8; ++r) acc[i][j][r] = 0.0f;

#ifndef USE_TDM
    const int ar = tid >> 1, ac = (tid & 1) * 16;  // 16 bf16 per thread per tile
#endif

    // ---- stage tile 0 ----
#ifdef USE_TDM
    if (wid == 0) {
        tdm_tile_load(&As[0][0][0], A + (size_t)blkM * K, K, M);
        tdm_tile_load(&Bs[0][0][0], Bt + (size_t)blkN * K, K, N);
    }
#else
    {
        const __bf16* ag = A + (size_t)(blkM + ar) * K + ac;
        *(v8bf*)&As[0][ar][ac]     = *(const v8bf*)ag;
        *(v8bf*)&As[0][ar][ac + 8] = *(const v8bf*)(ag + 8);
        const __bf16* bg = Bt + (size_t)(blkN + ar) * K + ac;
        *(v8bf*)&Bs[0][ar][ac]     = *(const v8bf*)bg;
        *(v8bf*)&Bs[0][ar][ac + 8] = *(const v8bf*)(bg + 8);
    }
#endif

    for (int it = 0; it < nk; ++it) {
        const int cur = it & 1;
        const int nxt = cur ^ 1;
        const int ktn = (it + 1) << 5;
#ifdef USE_TDM
        if (wid == 0) {
            if (it + 1 < nk) {
                tdm_tile_load(&As[nxt][0][0], A + (size_t)blkM * K + ktn, K, M);
                tdm_tile_load(&Bs[nxt][0][0], Bt + (size_t)blkN * K + ktn, K, N);
                __builtin_amdgcn_s_wait_tensorcnt(2);  // current pair complete
            } else {
                __builtin_amdgcn_s_wait_tensorcnt(0);
            }
        }
#else
        if (it + 1 < nk) {
            const __bf16* ag = A + (size_t)(blkM + ar) * K + ktn + ac;
            *(v8bf*)&As[nxt][ar][ac]     = *(const v8bf*)ag;
            *(v8bf*)&As[nxt][ar][ac + 8] = *(const v8bf*)(ag + 8);
            const __bf16* bg = Bt + (size_t)(blkN + ar) * K + ktn + ac;
            *(v8bf*)&Bs[nxt][ar][ac]     = *(const v8bf*)bg;
            *(v8bf*)&Bs[nxt][ar][ac + 8] = *(const v8bf*)(bg + 8);
        }
#endif
        __syncthreads();  // tile `cur` visible to all waves

        v16bf af[2], bfr[4];
        af[0] = frag_A(&As[cur][wm + l16][0], hi);
        af[1] = frag_A(&As[cur][wm + 16 + l16][0], hi);
#pragma unroll
        for (int j = 0; j < 4; ++j)
            bfr[j] = frag_B(&Bs[cur][wn + j * 16 + l16][0], hi);
#pragma unroll
        for (int i = 0; i < 2; ++i)
#pragma unroll
            for (int j = 0; j < 4; ++j) acc[i][j] = mma(af[i], bfr[j], acc[i][j]);
        __syncthreads();  // all reads of `cur` done before it is restaged
    }

    // ---- epilogue ----
#pragma unroll
    for (int i = 0; i < 2; ++i) {
#pragma unroll
        for (int j = 0; j < 4; ++j) {
            const int gn = blkN + wn + j * 16 + l16;
            const float bv = bias[gn];
#pragma unroll
            for (int r = 0; r < 8; ++r) {
                const int gm = blkM + wm + i * 16 + r + 8 * hi;
                float v = acc[i][j][r] + bv;
                if (MODE == 0) {
                    const int b = gm >> 7, s = gm & 127;
                    if (gn < HH) {
                        const int h = gn >> 6, d = gn & 63;
                        Qb[(((size_t)(b * NHH + h)) * SS + s) * DHH + d] =
                            (__bf16)(v * 0.125f);  // fold 1/sqrt(64)
                    } else if (gn < 2 * HH) {
                        const int n2 = gn - HH, h = n2 >> 6, d = n2 & 63;
                        Kb[(((size_t)(b * NHH + h)) * SS + s) * DHH + d] = (__bf16)v;
                    } else {
                        const int n2 = gn - 2 * HH, h = n2 >> 6, d = n2 & 63;
                        Vt[(((size_t)(b * NHH + h)) * DHH + d) * SS + s] = (__bf16)v;
                    }
                } else if (MODE == 1) {
                    outF[(size_t)gm * N + gn] = v;
                } else {
                    const float g = 0.5f * v * (1.0f + erff(v * 0.70710678118f));
                    outH[(size_t)gm * N + gn] = (__bf16)g;
                }
            }
        }
    }
}

// ---------------------------------------------------------------------------
// Fused attention per (b,h): scores = QK^T (Q pre-scaled) + mask bias,
// softmax fp32, ctx = P @ V. WMMA for both GEMMs. 384 blocks, 256 threads.
// ---------------------------------------------------------------------------
__global__ __launch_bounds__(256) void attn_kernel(
    const __bf16* __restrict__ Qg, const __bf16* __restrict__ Kg,
    const __bf16* __restrict__ Vg, const int* __restrict__ mask,
    __bf16* __restrict__ ctx) {
    __shared__ __align__(16) unsigned char raw[36864];  // Q+K, later P
    __shared__ __align__(16) __bf16 vt[64][136];
    __shared__ float bias_l[128];
    __bf16(*Qs)[72]  = (__bf16(*)[72])raw;
    __bf16(*Ks)[72]  = (__bf16(*)[72])(raw + 128 * 72 * 2);
    __bf16(*Ps)[136] = (__bf16(*)[136])raw;

    const int bh = blockIdx.x, b = bh / NHH, h = bh % NHH;
    const int tid = threadIdx.x, lane = tid & 31, wid = tid >> 5;
    const int hi = lane >> 4, l16 = lane & 15;

    const __bf16* qp = Qg + (size_t)bh * SS * DHH;
    const __bf16* kp = Kg + (size_t)bh * SS * DHH;
    const __bf16* vp = Vg + (size_t)bh * DHH * SS;

    {   // load Q, K (row-major [s][d])
        const int s = tid >> 1, c0 = (tid & 1) * 32;
#pragma unroll
        for (int j = 0; j < 32; j += 8) {
            *(v8bf*)&Qs[s][c0 + j] = *(const v8bf*)(qp + s * DHH + c0 + j);
            *(v8bf*)&Ks[s][c0 + j] = *(const v8bf*)(kp + s * DHH + c0 + j);
        }
    }
    {   // load V^T ([d][s])
        const int d = tid >> 2, c0 = (tid & 3) * 32;
#pragma unroll
        for (int j = 0; j < 32; j += 8)
            *(v8bf*)&vt[d][c0 + j] = *(const v8bf*)(vp + d * SS + c0 + j);
    }
    if (tid < 128) bias_l[tid] = (1.0f - (float)mask[b * SS + tid]) * -1.0e9f;
    __syncthreads();

    // ---- stage 1: scores = Q @ K^T  (M=128, N=128, K=64) ----
    const int wm = (wid & 3) * 32, wn = (wid >> 2) * 64;
    v8f acc1[2][4];
#pragma unroll
    for (int i = 0; i < 2; ++i)
#pragma unroll
        for (int j = 0; j < 4; ++j)
#pragma unroll
            for (int r = 0; r < 8; ++r) acc1[i][j][r] = 0.0f;

#pragma unroll
    for (int kt = 0; kt < 64; kt += 32) {
        v16bf af[2], bfr[4];
        af[0] = frag_A(&Qs[wm + l16][kt], hi);
        af[1] = frag_A(&Qs[wm + 16 + l16][kt], hi);
#pragma unroll
        for (int j = 0; j < 4; ++j) bfr[j] = frag_B(&Ks[wn + j * 16 + l16][kt], hi);
#pragma unroll
        for (int i = 0; i < 2; ++i)
#pragma unroll
            for (int j = 0; j < 4; ++j) acc1[i][j] = mma(af[i], bfr[j], acc1[i][j]);
    }
    __syncthreads();  // Q/K reads done; raw[] becomes P

#pragma unroll
    for (int i = 0; i < 2; ++i)
#pragma unroll
        for (int j = 0; j < 4; ++j)
#pragma unroll
            for (int r = 0; r < 8; ++r)
                Ps[wm + i * 16 + r + 8 * hi][wn + j * 16 + l16] = (__bf16)acc1[i][j][r];
    __syncthreads();

    // ---- softmax (fp32, one row per thread) ----
    if (tid < 128) {
        const int row = tid;
        float mx = -3.4e38f;
        for (int c = 0; c < 128; ++c)
            mx = fmaxf(mx, (float)Ps[row][c] + bias_l[c]);
        float sum = 0.0f;
        for (int c = 0; c < 128; ++c) {
            const float e = expf((float)Ps[row][c] + bias_l[c] - mx);
            sum += e;
            Ps[row][c] = (__bf16)e;
        }
        const float inv = 1.0f / sum;
        for (int c = 0; c < 128; ++c)
            Ps[row][c] = (__bf16)((float)Ps[row][c] * inv);
    }
    __syncthreads();

    // ---- stage 2: ctx = P @ V  (M=128, N=64, K=128) ----
    const int wm2 = (wid & 3) * 32, wn2 = (wid >> 2) * 32;
    v8f acc2[2][2];
#pragma unroll
    for (int i = 0; i < 2; ++i)
#pragma unroll
        for (int j = 0; j < 2; ++j)
#pragma unroll
            for (int r = 0; r < 8; ++r) acc2[i][j][r] = 0.0f;

#pragma unroll
    for (int kt = 0; kt < 128; kt += 32) {
        v16bf af[2], bfr[2];
        af[0]  = frag_A(&Ps[wm2 + l16][kt], hi);
        af[1]  = frag_A(&Ps[wm2 + 16 + l16][kt], hi);
        bfr[0] = frag_B(&vt[wn2 + l16][kt], hi);
        bfr[1] = frag_B(&vt[wn2 + 16 + l16][kt], hi);
#pragma unroll
        for (int i = 0; i < 2; ++i)
#pragma unroll
            for (int j = 0; j < 2; ++j) acc2[i][j] = mma(af[i], bfr[j], acc2[i][j]);
    }

#pragma unroll
    for (int i = 0; i < 2; ++i)
#pragma unroll
        for (int j = 0; j < 2; ++j)
#pragma unroll
            for (int r = 0; r < 8; ++r) {
                const int srow  = wm2 + i * 16 + r + 8 * hi;
                const int token = b * SS + srow;
                const int col   = h * DHH + wn2 + j * 16 + l16;
                ctx[(size_t)token * HH + col] = (__bf16)acc2[i][j][r];
            }
}

// ---------------------------------------------------------------------------
// Embedding + LayerNorm (one token per block)
// ---------------------------------------------------------------------------
__global__ __launch_bounds__(256) void embed_ln_kernel(
    const int* __restrict__ ids, const float* __restrict__ wemb,
    const float* __restrict__ pemb, const float* __restrict__ g,
    const float* __restrict__ be, float* __restrict__ xf, __bf16* __restrict__ xh) {
    __shared__ float xs[HH];
    __shared__ float red[256];
    __shared__ float s_mean, s_rstd;
    const int token = blockIdx.x, s = token & (SS - 1), tid = threadIdx.x;
    const int id = ids[token];
    float lsum = 0.0f;
    for (int i = tid; i < HH; i += 256) {
        const float v = wemb[(size_t)id * HH + i] + pemb[s * HH + i];
        xs[i] = v;
        lsum += v;
    }
    red[tid] = lsum;
    __syncthreads();
    for (int o = 128; o > 0; o >>= 1) {
        if (tid < o) red[tid] += red[tid + o];
        __syncthreads();
    }
    if (tid == 0) s_mean = red[0] * (1.0f / HH);
    __syncthreads();
    const float m = s_mean;
    float lv = 0.0f;
    for (int i = tid; i < HH; i += 256) { const float d = xs[i] - m; lv += d * d; }
    red[tid] = lv;
    __syncthreads();
    for (int o = 128; o > 0; o >>= 1) {
        if (tid < o) red[tid] += red[tid + o];
        __syncthreads();
    }
    if (tid == 0) s_rstd = rsqrtf(red[0] * (1.0f / HH) + LN_EPS);
    __syncthreads();
    const float r = s_rstd;
    for (int i = tid; i < HH; i += 256) {
        const float o = (xs[i] - m) * r * g[i] + be[i];
        xf[(size_t)token * HH + i] = o;
        xh[(size_t)token * HH + i] = (__bf16)o;
    }
}

// ---------------------------------------------------------------------------
// Residual add + LayerNorm: x = LN(x + y) (in place), also writes bf16 copy
// ---------------------------------------------------------------------------
__global__ __launch_bounds__(256) void add_ln_kernel(
    float* __restrict__ xf, const float* __restrict__ y,
    const float* __restrict__ g, const float* __restrict__ be,
    __bf16* __restrict__ xh) {
    __shared__ float xs[HH];
    __shared__ float red[256];
    __shared__ float s_mean, s_rstd;
    const int token = blockIdx.x, tid = threadIdx.x;
    float lsum = 0.0f;
    for (int i = tid; i < HH; i += 256) {
        const float v = xf[(size_t)token * HH + i] + y[(size_t)token * HH + i];
        xs[i] = v;
        lsum += v;
    }
    red[tid] = lsum;
    __syncthreads();
    for (int o = 128; o > 0; o >>= 1) {
        if (tid < o) red[tid] += red[tid + o];
        __syncthreads();
    }
    if (tid == 0) s_mean = red[0] * (1.0f / HH);
    __syncthreads();
    const float m = s_mean;
    float lv = 0.0f;
    for (int i = tid; i < HH; i += 256) { const float d = xs[i] - m; lv += d * d; }
    red[tid] = lv;
    __syncthreads();
    for (int o = 128; o > 0; o >>= 1) {
        if (tid < o) red[tid] += red[tid + o];
        __syncthreads();
    }
    if (tid == 0) s_rstd = rsqrtf(red[0] * (1.0f / HH) + LN_EPS);
    __syncthreads();
    const float r = s_rstd;
    for (int i = tid; i < HH; i += 256) {
        const float o = (xs[i] - m) * r * g[i] + be[i];
        xf[(size_t)token * HH + i] = o;
        xh[(size_t)token * HH + i] = (__bf16)o;
    }
}

// ---------------------------------------------------------------------------
// Classifier: emissions[token, t] = x @ cls_w + cls_b   (N=9, scalar fp32)
// ---------------------------------------------------------------------------
__global__ __launch_bounds__(256) void cls_kernel(const float* __restrict__ xf,
                                                  const float* __restrict__ w,
                                                  const float* __restrict__ bb,
                                                  float* __restrict__ em) {
    const int idx = blockIdx.x * 256 + threadIdx.x;
    if (idx >= NTOK * TT) return;
    const int token = idx / TT, t = idx % TT;
    float s = bb[t];
    const float* xr = xf + (size_t)token * HH;
    for (int k = 0; k < HH; ++k) s += xr[k] * w[k * TT + t];
    em[idx] = s;
}

// ---------------------------------------------------------------------------
// CRF NLL per batch (forward algorithm + gold score), 32 blocks x 32 threads
// ---------------------------------------------------------------------------
__global__ __launch_bounds__(32) void crf_kernel(
    const float* __restrict__ em, const int* __restrict__ labels,
    const int* __restrict__ mask, const float* __restrict__ start,
    const float* __restrict__ trans, const float* __restrict__ endv,
    float* __restrict__ outB) {
    __shared__ float tr[TT * TT];
    __shared__ float alpha[TT], alpha2[TT];
    const int b = blockIdx.x, tid = threadIdx.x;
    for (int i = tid; i < TT * TT; i += 32) tr[i] = trans[i];
    if (tid < TT) alpha[tid] = start[tid] + em[(size_t)(b * SS) * TT + tid];
    __syncthreads();
    for (int t = 1; t < SS; ++t) {
        float nxt = 0.0f;
        if (tid < TT) {
            float mx = -3.4e38f;
#pragma unroll
            for (int i = 0; i < TT; ++i) mx = fmaxf(mx, alpha[i] + tr[i * TT + tid]);
            float sum = 0.0f;
#pragma unroll
            for (int i = 0; i < TT; ++i) sum += expf(alpha[i] + tr[i * TT + tid] - mx);
            nxt = mx + logf(sum) + em[(size_t)(b * SS + t) * TT + tid];
            if (!(mask[b * SS + t] > 0)) nxt = alpha[tid];
        }
        __syncthreads();
        if (tid < TT) alpha2[tid] = nxt;
        __syncthreads();
        if (tid < TT) alpha[tid] = alpha2[tid];
        __syncthreads();
    }
    if (tid == 0) {
        float mx = -3.4e38f;
        for (int j = 0; j < TT; ++j) mx = fmaxf(mx, alpha[j] + endv[j]);
        float sum = 0.0f;
        for (int j = 0; j < TT; ++j) sum += expf(alpha[j] + endv[j] - mx);
        const float logz = mx + logf(sum);

        const int l0 = labels[b * SS];
        float sc = start[l0] + em[(size_t)(b * SS) * TT + l0];
        int prev = l0;
        int msum = (mask[b * SS] > 0) ? 1 : 0;
        for (int t = 1; t < SS; ++t) {
            const int lt = labels[b * SS + t];
            const float mf = (float)mask[b * SS + t];
            sc += mf * (tr[prev * TT + lt] + em[(size_t)(b * SS + t) * TT + lt]);
            msum += (mask[b * SS + t] > 0) ? 1 : 0;
            prev = lt;
        }
        const int last_idx = msum - 1;
        sc += endv[labels[b * SS + last_idx]];
        outB[b] = logz - sc;  // = -(score - log_z)
    }
}

__global__ __launch_bounds__(32) void nll_reduce_kernel(const float* __restrict__ pb,
                                                        float* __restrict__ out) {
    float v = pb[threadIdx.x];
    for (int o = 16; o > 0; o >>= 1) v += __shfl_down(v, o, 32);
    if (threadIdx.x == 0) out[0] = v * (1.0f / BB);
}

// ---------------------------------------------------------------------------
// Host driver
// ---------------------------------------------------------------------------
extern "C" void kernel_launch(void* const* d_in, const int* in_sizes, int n_in,
                              void* d_out, int out_size, void* d_ws, size_t ws_size,
                              hipStream_t stream) {
    (void)in_sizes; (void)n_in; (void)out_size; (void)ws_size;

    const int*   input_ids = (const int*)d_in[0];
    const int*   attn_mask = (const int*)d_in[1];
    const int*   labels    = (const int*)d_in[2];
    const float* word_emb  = (const float*)d_in[3];
    const float* pos_emb   = (const float*)d_in[4];
    const float* emb_ln_s  = (const float*)d_in[5];
    const float* emb_ln_b  = (const float*)d_in[6];
    const float* qkv_w     = (const float*)d_in[7];
    const float* qkv_b     = (const float*)d_in[8];
    const float* out_w     = (const float*)d_in[9];
    const float* out_b     = (const float*)d_in[10];
    const float* ln1_s     = (const float*)d_in[11];
    const float* ln1_b     = (const float*)d_in[12];
    const float* ffn_w1    = (const float*)d_in[13];
    const float* ffn_b1    = (const float*)d_in[14];
    const float* ffn_w2    = (const float*)d_in[15];
    const float* ffn_b2    = (const float*)d_in[16];
    const float* ln2_s     = (const float*)d_in[17];
    const float* ln2_b     = (const float*)d_in[18];
    const float* cls_w     = (const float*)d_in[19];
    const float* cls_b     = (const float*)d_in[20];
    const float* crf_start = (const float*)d_in[21];
    const float* crf_trans = (const float*)d_in[22];
    const float* crf_end   = (const float*)d_in[23];

    char* ws = (char*)d_ws;
    size_t off = 0;
    auto take = [&](size_t bytes) -> char* {
        char* p = ws + off;
        off += (bytes + 255) & ~(size_t)255;
        return p;
    };
    float*  xf   = (float*)take((size_t)NTOK * HH * 4);
    __bf16* xh   = (__bf16*)take((size_t)NTOK * HH * 2);
    __bf16* Qb   = (__bf16*)take((size_t)BB * NHH * SS * DHH * 2);
    __bf16* Kb   = (__bf16*)take((size_t)BB * NHH * SS * DHH * 2);
    __bf16* Vt   = (__bf16*)take((size_t)BB * NHH * SS * DHH * 2);
    __bf16* ctx  = (__bf16*)take((size_t)NTOK * HH * 2);
    float*  tmp  = (float*)take((size_t)NTOK * HH * 4);
    __bf16* hb   = (__bf16*)take((size_t)NTOK * FFF * 2);
    float*  em   = (float*)take((size_t)NTOK * TT * 4);
    float*  pb   = (float*)take((size_t)BB * 4);
    __bf16* qkvT = (__bf16*)take((size_t)LL * 3 * HH * HH * 2);  // [N=2304][K=768]
    __bf16* outT = (__bf16*)take((size_t)LL * HH * HH * 2);      // [768][768]
    __bf16* w1T  = (__bf16*)take((size_t)LL * FFF * HH * 2);     // [3072][768]
    __bf16* w2T  = (__bf16*)take((size_t)LL * HH * FFF * 2);     // [768][3072]

    // ---- weight pre-pass: fp32 [K,N] -> bf16 transposed [N,K] ----
    for (int l = 0; l < LL; ++l) {
        wt_kernel<<<dim3(3 * HH / 32, HH / 32), 256, 0, stream>>>(
            qkv_w + (size_t)l * HH * 3 * HH, qkvT + (size_t)l * 3 * HH * HH, HH,
            3 * HH);
        wt_kernel<<<dim3(HH / 32, HH / 32), 256, 0, stream>>>(
            out_w + (size_t)l * HH * HH, outT + (size_t)l * HH * HH, HH, HH);
        wt_kernel<<<dim3(FFF / 32, HH / 32), 256, 0, stream>>>(
            ffn_w1 + (size_t)l * HH * FFF, w1T + (size_t)l * FFF * HH, HH, FFF);
        wt_kernel<<<dim3(HH / 32, FFF / 32), 256, 0, stream>>>(
            ffn_w2 + (size_t)l * FFF * HH, w2T + (size_t)l * HH * FFF, FFF, HH);
    }

    // ---- embedding + LN ----
    embed_ln_kernel<<<NTOK, 256, 0, stream>>>(input_ids, word_emb, pos_emb,
                                              emb_ln_s, emb_ln_b, xf, xh);

    for (int l = 0; l < LL; ++l) {
        const __bf16* qw = qkvT + (size_t)l * 3 * HH * HH;
        const float*  qb = qkv_b + (size_t)l * 3 * HH;
        const __bf16* ow = outT + (size_t)l * HH * HH;
        const float*  ob = out_b + (size_t)l * HH;
        const __bf16* w1 = w1T + (size_t)l * FFF * HH;
        const float*  b1 = ffn_b1 + (size_t)l * FFF;
        const __bf16* w2 = w2T + (size_t)l * HH * FFF;
        const float*  b2 = ffn_b2 + (size_t)l * HH;

        gemm_kernel<0><<<dim3(3 * HH / 128, NTOK / 128), 256, 0, stream>>>(
            xh, qw, qb, nullptr, nullptr, Qb, Kb, Vt, NTOK, 3 * HH, HH);
        attn_kernel<<<BB * NHH, 256, 0, stream>>>(Qb, Kb, Vt, attn_mask, ctx);
        gemm_kernel<1><<<dim3(HH / 128, NTOK / 128), 256, 0, stream>>>(
            ctx, ow, ob, tmp, nullptr, nullptr, nullptr, nullptr, NTOK, HH, HH);
        add_ln_kernel<<<NTOK, 256, 0, stream>>>(xf, tmp, ln1_s + (size_t)l * HH,
                                                ln1_b + (size_t)l * HH, xh);
        gemm_kernel<2><<<dim3(FFF / 128, NTOK / 128), 256, 0, stream>>>(
            xh, w1, b1, nullptr, hb, nullptr, nullptr, nullptr, NTOK, FFF, HH);
        gemm_kernel<1><<<dim3(HH / 128, NTOK / 128), 256, 0, stream>>>(
            hb, w2, b2, tmp, nullptr, nullptr, nullptr, nullptr, NTOK, HH, FFF);
        add_ln_kernel<<<NTOK, 256, 0, stream>>>(xf, tmp, ln2_s + (size_t)l * HH,
                                                ln2_b + (size_t)l * HH, xh);
    }

    cls_kernel<<<(NTOK * TT + 255) / 256, 256, 0, stream>>>(xf, cls_w, cls_b, em);
    crf_kernel<<<BB, 32, 0, stream>>>(em, labels, attn_mask, crf_start, crf_trans,
                                      crf_end, pb);
    nll_reduce_kernel<<<1, 32, 0, stream>>>(pb, (float*)d_out);
}
